// RNNModelWithSkipConnections_28174985462581
// MI455X (gfx1250) — compile-verified, matrix operand order checked by reading, not run on hardware
//
#include <hip/hip_runtime.h>
#include <hip/hip_bf16.h>

// ---------------- types ----------------
typedef __attribute__((ext_vector_type(16))) _Float16 v16h;
typedef __attribute__((ext_vector_type(8)))  _Float16 v8h;
typedef __attribute__((ext_vector_type(8)))  float    v8f;
using half_t = _Float16;

// ---------------- model constants ----------------
constexpr int Bn    = 64;     // batch
constexpr int Tn    = 256;    // time steps
constexpr int NIN   = 161;
constexpr int Hn    = 1024;
constexpr int NOUT  = 161;
constexpr int NOUTP = 176;    // padded to 16
constexpr int Fc    = 64;
constexpr int KF    = 21;
constexpr int KTc   = 11;
constexpr int CONVH = 71;
constexpr int RNNIN = Fc * CONVH;   // 4544
constexpr int K0P   = 192;          // 161 padded to mult of 32
constexpr int TB    = Tn * Bn;      // 16384

// ---------------- WMMA helpers ----------------
// 16-bit A/B fragment, per CDNA5 ISA layout:
// lanes 0-15: halves 0-7 -> K 0..7,  halves 8-15 -> K 16..23 (row = lane&15)
// lanes16-31: halves 0-7 -> K 8..15, halves 8-15 -> K 24..31
__device__ __forceinline__ v16h ldfrag(const half_t* p) {
  v8h lo = *reinterpret_cast<const v8h*>(p);
  v8h hi = *reinterpret_cast<const v8h*>(p + 16);
  v16h r;
#pragma unroll
  for (int i = 0; i < 8; ++i) { r[i] = lo[i]; r[i + 8] = hi[i]; }
  return r;
}

__device__ __forceinline__ v8f wmma32(v16h a, v16h b, v8f c) {
  return __builtin_amdgcn_wmma_f32_16x16x32_f16(false, a, false, b, (short)0, c,
                                                false, false);
}

// one 16x16 output tile of  C = A(16xK) * W(16 rows of len K)^T, f32 acc
__device__ __forceinline__ v8f gemm_tile(const half_t* A, int lda,
                                         const half_t* W, int ldw, int K) {
  const int lane = threadIdx.x & 31;
  const int rsel = lane & 15;
  const int koff = (lane >> 4) * 8;
  const half_t* pa = A + (size_t)rsel * lda + koff;
  const half_t* pw = W + (size_t)rsel * ldw + koff;
  v8f acc = {};
#pragma unroll 4
  for (int k = 0; k < K; k += 32)
    acc = wmma32(ldfrag(pa + k), ldfrag(pw + k), acc);
  return acc;
}

__device__ __forceinline__ float sigm(float x) {
  return 1.0f / (1.0f + __expf(-x));
}

// ---------------- prep kernels ----------------
__global__ void k_cvt(const float* __restrict__ s, half_t* __restrict__ d, int n) {
  int i = blockIdx.x * 256 + threadIdx.x;
  if (i < n) d[i] = (half_t)s[i];
}

__global__ void k_wout(const float* __restrict__ w, half_t* __restrict__ d) {
  int i = blockIdx.x * 256 + threadIdx.x;      // NOUTP*1024
  if (i >= NOUTP * Hn) return;
  int n = i / Hn, k = i % Hn;
  d[i] = (n < NOUT) ? (half_t)w[n * Hn + k] : (half_t)0.0f;
}

__global__ void k_winpad(const float* __restrict__ w, half_t* __restrict__ d) {
  int i = blockIdx.x * 256 + threadIdx.x;      // 1024*192
  if (i >= Hn * K0P) return;
  int n = i / K0P, k = i % K0P;
  d[i] = (k < NIN) ? (half_t)w[n * NIN + k] : (half_t)0.0f;
}

__global__ void k_x0(const float* __restrict__ x, half_t* __restrict__ d) {
  int i = blockIdx.x * 256 + threadIdx.x;      // TB*192
  if (i >= TB * K0P) return;
  int k = i % K0P, row = i / K0P;
  int b = row & 63, t = row >> 6;
  d[i] = (k < NIN) ? (half_t)x[(size_t)b * NIN * Tn + (size_t)k * Tn + t]
                   : (half_t)0.0f;
}

__global__ void k_hinit(const float* __restrict__ h, float* __restrict__ hf,
                        half_t* __restrict__ hh) {
  int i = blockIdx.x * 256 + threadIdx.x;
  if (i >= Bn * Hn) return;
  float v = h[i];
  hf[i] = v;
  hh[i] = (half_t)v;
}

// conv2d(1->64, 21x11, stride (2,1), pad (0,5)) -> x_rnn f16 [TB, 4544]
__global__ void k_conv(const float* __restrict__ x, const float* __restrict__ cw,
                       const float* __restrict__ cb, half_t* __restrict__ xr) {
  size_t i = (size_t)blockIdx.x * 256 + threadIdx.x;
  if (i >= (size_t)TB * RNNIN) return;
  int col = (int)(i % RNNIN);
  int row = (int)(i / RNNIN);
  int b = row & 63, t = row >> 6;
  int f = col / CONVH, hh = col % CONVH;
  float acc = cb[f];
  const float* px = x + (size_t)b * NIN * Tn;
  const float* pw = cw + (size_t)f * KF * KTc;
#pragma unroll 3
  for (int kf = 0; kf < KF; ++kf) {
    const float* pr = px + (size_t)(2 * hh + kf) * Tn;
    const float* pk = pw + kf * KTc;
#pragma unroll
    for (int kt = 0; kt < KTc; ++kt) {
      int tt = t + kt - 5;
      if (tt >= 0 && tt < Tn) acc = fmaf(pr[tt], pk[kt], acc);
    }
  }
  xr[i] = (half_t)acc;
}

// x2/x3 projections: [TB,192] @ [1024,192]^T + bias -> f16 [TB,1024]
__global__ void k_gemm_in(const half_t* __restrict__ X, const half_t* __restrict__ W,
                          const float* __restrict__ bias, half_t* __restrict__ out) {
  int wid = threadIdx.x >> 5;
  int gid = blockIdx.x * 8 + wid;         // 65536 waves
  int mt = gid >> 6, ntl = gid & 63;
  v8f acc = gemm_tile(X + (size_t)mt * 16 * K0P, K0P,
                      W + (size_t)ntl * 16 * K0P, K0P, K0P);
  int lane = threadIdx.x & 31;
  int nl = ntl * 16 + (lane & 15);
  float bb = bias[nl];
#pragma unroll
  for (int r = 0; r < 8; ++r) {
    int row = mt * 16 + (lane >> 4) * 8 + r;
    out[(size_t)row * Hn + nl] = (half_t)(acc[r] + bb);
  }
}

// xg0 = x_rnn[TB,4544] @ W_ih0[3072,4544]^T -> f16 [TB,3072]; wave does 16x32
__global__ void k_gemm_xg0(const half_t* __restrict__ X, const half_t* __restrict__ W,
                           half_t* __restrict__ out) {
  int wid = threadIdx.x >> 5;
  int gid = blockIdx.x * 8 + wid;         // 98304 waves
  int mt = gid / 96, np = gid % 96;
  int n0 = np * 32;
  const int lane = threadIdx.x & 31;
  const int rsel = lane & 15, koff = (lane >> 4) * 8;
  const half_t* pa = X + ((size_t)(mt * 16 + rsel)) * RNNIN + koff;
  const half_t* pb0 = W + ((size_t)(n0 + rsel)) * RNNIN + koff;
  const half_t* pb1 = W + ((size_t)(n0 + 16 + rsel)) * RNNIN + koff;
  v8f a0 = {}, a1 = {};
#pragma unroll 2
  for (int k = 0; k < RNNIN; k += 32) {
    v16h a = ldfrag(pa + k);
    a0 = wmma32(a, ldfrag(pb0 + k), a0);
    a1 = wmma32(a, ldfrag(pb1 + k), a1);
  }
  int nl = lane & 15;
#pragma unroll
  for (int r = 0; r < 8; ++r) {
    int row = mt * 16 + (lane >> 4) * 8 + r;
    out[(size_t)row * 3072 + n0 + nl] = (half_t)a0[r];
    out[(size_t)row * 3072 + n0 + 16 + nl] = (half_t)a1[r];
  }
}

// ---------------- per-timestep kernels ----------------
// GRU layer 1: hg = h1_old @ W_hh0^T (3 gates), combine with precomputed xg0[t]
__global__ void k_gru1(const half_t* __restrict__ hOld, const half_t* __restrict__ Whh,
                       const half_t* __restrict__ xg0, float* __restrict__ hf,
                       half_t* __restrict__ hNew, int t) {
  int wid = threadIdx.x >> 5;
  int gid = blockIdx.x * 8 + wid;         // 256 waves
  int mt = gid >> 6, jt = gid & 63;
  const int lane = threadIdx.x & 31;
  const int rsel = lane & 15, koff = (lane >> 4) * 8;
  const half_t* pa = hOld + ((size_t)(mt * 16 + rsel)) * Hn + koff;
  const half_t* pr = Whh + ((size_t)(0 * Hn + jt * 16 + rsel)) * Hn + koff;
  const half_t* pz = Whh + ((size_t)(1 * Hn + jt * 16 + rsel)) * Hn + koff;
  const half_t* pn = Whh + ((size_t)(2 * Hn + jt * 16 + rsel)) * Hn + koff;
  v8f ar = {}, az = {}, an = {};
#pragma unroll 2
  for (int k = 0; k < Hn; k += 32) {
    v16h a = ldfrag(pa + k);
    ar = wmma32(a, ldfrag(pr + k), ar);
    az = wmma32(a, ldfrag(pz + k), az);
    an = wmma32(a, ldfrag(pn + k), an);
  }
  int nl = jt * 16 + (lane & 15);
#pragma unroll
  for (int r = 0; r < 8; ++r) {
    int row = mt * 16 + (lane >> 4) * 8 + r;
    const half_t* xg = xg0 + (size_t)(t * Bn + row) * 3072;
    float rg = sigm((float)xg[nl] + ar[r]);
    float zg = sigm((float)xg[nl + Hn] + az[r]);
    float ng = tanhf((float)xg[nl + 2 * Hn] + rg * an[r]);
    size_t i = (size_t)row * Hn + nl;
    float hv = (1.0f - zg) * ng + zg * hf[i];
    hf[i] = hv;
    hNew[i] = (half_t)hv;
  }
}

// h1' projections through ht0 / ht1 / ot0 (3 weight sections of N=1024 each)
__global__ void k_trans1(const half_t* __restrict__ h1, const half_t* __restrict__ Wa,
                         const half_t* __restrict__ Wb, const half_t* __restrict__ Wc,
                         const float* __restrict__ ba, const float* __restrict__ bb,
                         const float* __restrict__ bc, const half_t* __restrict__ x2all,
                         half_t* __restrict__ u2h, half_t* __restrict__ t12h,
                         float* __restrict__ t13f, float* __restrict__ rbuf, int t) {
  int wid = threadIdx.x >> 5;
  int gid = blockIdx.x * 8 + wid;         // 768 waves
  int s = gid >> 8, rem = gid & 255;
  int mt = rem >> 6, ntl = rem & 63;
  const half_t* W = (s == 0) ? Wa : (s == 1) ? Wb : Wc;
  const float* bias = (s == 0) ? ba : (s == 1) ? bb : bc;
  v8f acc = gemm_tile(h1 + (size_t)mt * 16 * Hn, Hn,
                      W + (size_t)ntl * 16 * Hn, Hn, Hn);
  int lane = threadIdx.x & 31;
  int nl = ntl * 16 + (lane & 15);
  float bv = bias[nl];
#pragma unroll
  for (int r = 0; r < 8; ++r) {
    int row = mt * 16 + (lane >> 4) * 8 + r;
    size_t i = (size_t)row * Hn + nl;
    float val = acc[r] + bv;
    if (s == 0) {
      float u = (float)x2all[(size_t)(t * Bn + row) * Hn + nl] + val;
      u2h[i] = (half_t)u;
      t12h[i] = (half_t)val;
    } else if (s == 1) {
      t13f[i] = val;
    } else {
      rbuf[i] = val;
    }
  }
}

// fused GRU (layers 2 & 3): r/z accumulate x- and h-gates together; n split.
__global__ void k_gru_fused(const half_t* __restrict__ A1, const half_t* __restrict__ Wx,
                            const half_t* __restrict__ A2, const half_t* __restrict__ Wh,
                            float* __restrict__ hf, half_t* __restrict__ hNew) {
  int wid = threadIdx.x >> 5;
  int gid = blockIdx.x * 8 + wid;         // 256 waves
  int mt = gid >> 6, jt = gid & 63;
  const int lane = threadIdx.x & 31;
  const int rsel = lane & 15, koff = (lane >> 4) * 8;
  const half_t* pa1 = A1 + ((size_t)(mt * 16 + rsel)) * Hn + koff;
  const half_t* pa2 = A2 + ((size_t)(mt * 16 + rsel)) * Hn + koff;
  const half_t* wx0 = Wx + ((size_t)(0 * Hn + jt * 16 + rsel)) * Hn + koff;
  const half_t* wx1 = Wx + ((size_t)(1 * Hn + jt * 16 + rsel)) * Hn + koff;
  const half_t* wx2 = Wx + ((size_t)(2 * Hn + jt * 16 + rsel)) * Hn + koff;
  const half_t* wh0 = Wh + ((size_t)(0 * Hn + jt * 16 + rsel)) * Hn + koff;
  const half_t* wh1 = Wh + ((size_t)(1 * Hn + jt * 16 + rsel)) * Hn + koff;
  const half_t* wh2 = Wh + ((size_t)(2 * Hn + jt * 16 + rsel)) * Hn + koff;
  v8f ar = {}, az = {}, axn = {}, ahn = {};
  for (int k = 0; k < Hn; k += 32) {
    v16h a1 = ldfrag(pa1 + k);
    v16h a2 = ldfrag(pa2 + k);
    ar = wmma32(a1, ldfrag(wx0 + k), ar);
    ar = wmma32(a2, ldfrag(wh0 + k), ar);
    az = wmma32(a1, ldfrag(wx1 + k), az);
    az = wmma32(a2, ldfrag(wh1 + k), az);
    axn = wmma32(a1, ldfrag(wx2 + k), axn);
    ahn = wmma32(a2, ldfrag(wh2 + k), ahn);
  }
  int nl = jt * 16 + (lane & 15);
#pragma unroll
  for (int r = 0; r < 8; ++r) {
    int row = mt * 16 + (lane >> 4) * 8 + r;
    size_t i = (size_t)row * Hn + nl;
    float rg = sigm(ar[r]);
    float zg = sigm(az[r]);
    float ng = tanhf(axn[r] + rg * ahn[r]);
    float hv = (1.0f - zg) * ng + zg * hf[i];
    hf[i] = hv;
    hNew[i] = (half_t)hv;
  }
}

// step4: sub0 = t12h@ht2^T -> u3 ; sub1 = h2'@ot1^T -> rbuf += r2
__global__ void k_step4(const half_t* __restrict__ t12h, const half_t* __restrict__ Wht2,
                        const float* __restrict__ bht2, const half_t* __restrict__ x3all,
                        const float* __restrict__ t13f, half_t* __restrict__ u3h,
                        const half_t* __restrict__ h2, const half_t* __restrict__ Wot1,
                        const float* __restrict__ bot1, float* __restrict__ rbuf, int t) {
  int wid = threadIdx.x >> 5;
  int gid = blockIdx.x * 8 + wid;         // 512 waves
  int sub = gid >> 8, rem = gid & 255;
  int mt = rem >> 6, ntl = rem & 63;
  const half_t* A = sub ? h2 : t12h;
  const half_t* W = sub ? Wot1 : Wht2;
  const float* bias = sub ? bot1 : bht2;
  v8f acc = gemm_tile(A + (size_t)mt * 16 * Hn, Hn,
                      W + (size_t)ntl * 16 * Hn, Hn, Hn);
  int lane = threadIdx.x & 31;
  int nl = ntl * 16 + (lane & 15);
  float bv = bias[nl];
#pragma unroll
  for (int r = 0; r < 8; ++r) {
    int row = mt * 16 + (lane >> 4) * 8 + r;
    size_t i = (size_t)row * Hn + nl;
    float val = acc[r] + bv;
    if (sub == 0) {
      float u = (float)x3all[(size_t)(t * Bn + row) * Hn + nl] + t13f[i] + val;
      u3h[i] = (half_t)u;
    } else {
      rbuf[i] = rbuf[i] + val;
    }
  }
}

// r3 = h3'@ot2^T + b; rtot = (r1+r2)+r3 -> f16
__global__ void k_r3(const half_t* __restrict__ h3, const half_t* __restrict__ Wot2,
                     const float* __restrict__ bot2, const float* __restrict__ rbuf,
                     half_t* __restrict__ rtot) {
  int wid = threadIdx.x >> 5;
  int gid = blockIdx.x * 8 + wid;         // 256 waves
  int mt = gid >> 6, ntl = gid & 63;
  v8f acc = gemm_tile(h3 + (size_t)mt * 16 * Hn, Hn,
                      Wot2 + (size_t)ntl * 16 * Hn, Hn, Hn);
  int lane = threadIdx.x & 31;
  int nl = ntl * 16 + (lane & 15);
  float bv = bot2[nl];
#pragma unroll
  for (int r = 0; r < 8; ++r) {
    int row = mt * 16 + (lane >> 4) * 8 + r;
    size_t i = (size_t)row * Hn + nl;
    rtot[i] = (half_t)(rbuf[i] + acc[r] + bv);
  }
}

// y_t = rtot @ out_W^T + b -> d_out[t,:,:]
__global__ void k_out(const half_t* __restrict__ rtot, const half_t* __restrict__ Wout,
                      const float* __restrict__ bout, float* __restrict__ out, int t) {
  int wid = threadIdx.x >> 5;
  int gid = blockIdx.x * 8 + wid;         // 48 waves, 44 used
  if (gid >= 44) return;                  // wave-uniform exit (before any WMMA)
  int mt = gid / 11, ntl = gid % 11;
  v8f acc = gemm_tile(rtot + (size_t)mt * 16 * Hn, Hn,
                      Wout + (size_t)ntl * 16 * Hn, Hn, Hn);
  int lane = threadIdx.x & 31;
  int nl = ntl * 16 + (lane & 15);
#pragma unroll
  for (int r = 0; r < 8; ++r) {
    int row = mt * 16 + (lane >> 4) * 8 + r;
    if (nl < NOUT)
      out[(size_t)t * Bn * NOUT + (size_t)row * NOUT + nl] = acc[r] + bout[nl];
  }
}

// ---------------- host launcher ----------------
extern "C" void kernel_launch(void* const* d_in, const int* in_sizes, int n_in,
                              void* d_out, int out_size, void* d_ws, size_t ws_size,
                              hipStream_t stream) {
  (void)in_sizes; (void)n_in; (void)out_size; (void)ws_size;
  const float* x     = (const float*)d_in[0];
  const float* h1i   = (const float*)d_in[1];
  const float* h2i   = (const float*)d_in[2];
  const float* h3i   = (const float*)d_in[3];
  const float* convW = (const float*)d_in[4];
  const float* convb = (const float*)d_in[5];
  const float* wih0  = (const float*)d_in[6];
  const float* whh0  = (const float*)d_in[7];
  const float* wih1  = (const float*)d_in[8];
  const float* whh1  = (const float*)d_in[9];
  const float* wih2  = (const float*)d_in[10];
  const float* whh2  = (const float*)d_in[11];
  const float* in1W  = (const float*)d_in[12];
  const float* in1b  = (const float*)d_in[13];
  const float* in2W  = (const float*)d_in[14];
  const float* in2b  = (const float*)d_in[15];
  const float* ht0W  = (const float*)d_in[16];
  const float* ht0b  = (const float*)d_in[17];
  const float* ht1W  = (const float*)d_in[18];
  const float* ht1b  = (const float*)d_in[19];
  const float* ht2W  = (const float*)d_in[20];
  const float* ht2b  = (const float*)d_in[21];
  const float* ot0W  = (const float*)d_in[22];
  const float* ot0b  = (const float*)d_in[23];
  const float* ot1W  = (const float*)d_in[24];
  const float* ot1b  = (const float*)d_in[25];
  const float* ot2W  = (const float*)d_in[26];
  const float* ot2b  = (const float*)d_in[27];
  const float* outW  = (const float*)d_in[28];
  const float* outb  = (const float*)d_in[29];

  // ---- workspace carve-out ----
  char* base = (char*)d_ws;
  size_t off = 0;
  auto alloc = [&](size_t bytes) -> void* {
    void* p = base + off;
    off += (bytes + 255) & ~(size_t)255;
    return p;
  };
  half_t* wih0h = (half_t*)alloc((size_t)3072 * RNNIN * 2);
  half_t* whh0h = (half_t*)alloc((size_t)3072 * Hn * 2);
  half_t* wih1h = (half_t*)alloc((size_t)3072 * Hn * 2);
  half_t* whh1h = (half_t*)alloc((size_t)3072 * Hn * 2);
  half_t* wih2h = (half_t*)alloc((size_t)3072 * Hn * 2);
  half_t* whh2h = (half_t*)alloc((size_t)3072 * Hn * 2);
  half_t* ht0h  = (half_t*)alloc((size_t)Hn * Hn * 2);
  half_t* ht1h  = (half_t*)alloc((size_t)Hn * Hn * 2);
  half_t* ht2h  = (half_t*)alloc((size_t)Hn * Hn * 2);
  half_t* ot0h  = (half_t*)alloc((size_t)Hn * Hn * 2);
  half_t* ot1h  = (half_t*)alloc((size_t)Hn * Hn * 2);
  half_t* ot2h  = (half_t*)alloc((size_t)Hn * Hn * 2);
  half_t* wouth = (half_t*)alloc((size_t)NOUTP * Hn * 2);
  half_t* win1h = (half_t*)alloc((size_t)Hn * K0P * 2);
  half_t* win2h = (half_t*)alloc((size_t)Hn * K0P * 2);
  half_t* x0h   = (half_t*)alloc((size_t)TB * K0P * 2);
  half_t* xrnnh = (half_t*)alloc((size_t)TB * RNNIN * 2);
  half_t* xg0h  = (half_t*)alloc((size_t)TB * 3072 * 2);
  half_t* x2all = (half_t*)alloc((size_t)TB * Hn * 2);
  half_t* x3all = (half_t*)alloc((size_t)TB * Hn * 2);
  float*  h1f   = (float*)alloc((size_t)Bn * Hn * 4);
  float*  h2f   = (float*)alloc((size_t)Bn * Hn * 4);
  float*  h3f   = (float*)alloc((size_t)Bn * Hn * 4);
  half_t* h1hb  = (half_t*)alloc((size_t)2 * Bn * Hn * 2);   // ping-pong
  half_t* h2hb  = (half_t*)alloc((size_t)2 * Bn * Hn * 2);
  half_t* h3hb  = (half_t*)alloc((size_t)2 * Bn * Hn * 2);
  half_t* t12h  = (half_t*)alloc((size_t)Bn * Hn * 2);
  float*  t13f  = (float*)alloc((size_t)Bn * Hn * 4);
  float*  rbuf  = (float*)alloc((size_t)Bn * Hn * 4);
  half_t* u2h   = (half_t*)alloc((size_t)Bn * Hn * 2);
  half_t* u3h   = (half_t*)alloc((size_t)Bn * Hn * 2);
  half_t* rtoth = (half_t*)alloc((size_t)Bn * Hn * 2);

  auto cvt = [&](const float* s, half_t* d, size_t n) {
    k_cvt<<<(int)((n + 255) / 256), 256, 0, stream>>>(s, d, (int)n);
  };

  // ---- weight / input preparation ----
  cvt(wih0, wih0h, (size_t)3072 * RNNIN);
  cvt(whh0, whh0h, (size_t)3072 * Hn);
  cvt(wih1, wih1h, (size_t)3072 * Hn);
  cvt(whh1, whh1h, (size_t)3072 * Hn);
  cvt(wih2, wih2h, (size_t)3072 * Hn);
  cvt(whh2, whh2h, (size_t)3072 * Hn);
  cvt(ht0W, ht0h, (size_t)Hn * Hn);
  cvt(ht1W, ht1h, (size_t)Hn * Hn);
  cvt(ht2W, ht2h, (size_t)Hn * Hn);
  cvt(ot0W, ot0h, (size_t)Hn * Hn);
  cvt(ot1W, ot1h, (size_t)Hn * Hn);
  cvt(ot2W, ot2h, (size_t)Hn * Hn);
  k_wout<<<(NOUTP * Hn + 255) / 256, 256, 0, stream>>>(outW, wouth);
  k_winpad<<<(Hn * K0P + 255) / 256, 256, 0, stream>>>(in1W, win1h);
  k_winpad<<<(Hn * K0P + 255) / 256, 256, 0, stream>>>(in2W, win2h);
  k_x0<<<(TB * K0P + 255) / 256, 256, 0, stream>>>(x, x0h);
  k_hinit<<<(Bn * Hn + 255) / 256, 256, 0, stream>>>(h1i, h1f, h1hb);
  k_hinit<<<(Bn * Hn + 255) / 256, 256, 0, stream>>>(h2i, h2f, h2hb);
  k_hinit<<<(Bn * Hn + 255) / 256, 256, 0, stream>>>(h3i, h3f, h3hb);
  k_conv<<<(int)(((size_t)TB * RNNIN + 255) / 256), 256, 0, stream>>>(x, convW, convb, xrnnh);

  // ---- time-parallel GEMMs ----
  k_gemm_in<<<8192, 256, 0, stream>>>(x0h, win1h, in1b, x2all);
  k_gemm_in<<<8192, 256, 0, stream>>>(x0h, win2h, in2b, x3all);
  k_gemm_xg0<<<12288, 256, 0, stream>>>(xrnnh, wih0h, xg0h);

  // ---- sequential scan ----
  for (int t = 0; t < Tn; ++t) {
    int po = t & 1, pn = po ^ 1;
    half_t* h1o = h1hb + (size_t)po * Bn * Hn;
    half_t* h1n = h1hb + (size_t)pn * Bn * Hn;
    half_t* h2o = h2hb + (size_t)po * Bn * Hn;
    half_t* h2n = h2hb + (size_t)pn * Bn * Hn;
    half_t* h3o = h3hb + (size_t)po * Bn * Hn;
    half_t* h3n = h3hb + (size_t)pn * Bn * Hn;

    k_gru1<<<32, 256, 0, stream>>>(h1o, whh0h, xg0h, h1f, h1n, t);
    k_trans1<<<96, 256, 0, stream>>>(h1n, ht0h, ht1h, ot0h, ht0b, ht1b, ot0b,
                                     x2all, u2h, t12h, t13f, rbuf, t);
    k_gru_fused<<<32, 256, 0, stream>>>(u2h, wih1h, h2o, whh1h, h2f, h2n);
    k_step4<<<64, 256, 0, stream>>>(t12h, ht2h, ht2b, x3all, t13f, u3h,
                                    h2n, ot1h, ot1b, rbuf, t);
    k_gru_fused<<<32, 256, 0, stream>>>(u3h, wih2h, h3o, whh2h, h3f, h3n);
    k_r3<<<32, 256, 0, stream>>>(h3n, ot2h, ot2b, rbuf, rtoth);
    k_out<<<6, 256, 0, stream>>>(rtoth, wouth, outb, (float*)d_out, t);
  }
}
